// EncoderLSTM_1606317769035
// MI455X (gfx1250) — matrix operand order repaired and checked by reading, not compiled, not run on hardware
//
#include <hip/hip_runtime.h>
#include <hip/hip_bf16.h>

// ---------------- problem constants ----------------
#define B_    128
#define T_    128
#define D_    300
#define H_    2048
#define FOURH 8192
#define KXP   320           // x part padded to multiple of 32
#define KPAD  2368          // KXP + H_
#define NCHUNK (KPAD / 32)  // 74 K-chunks of 32
#define XCHUNK (KXP / 32)   // 10 chunks come from x, rest from h
#define BROW  40            // LDS K-row stride (elements): 80B, 16B-aligned, conflict-free

typedef __attribute__((ext_vector_type(16))) __bf16    v16bf;
typedef __attribute__((ext_vector_type(8)))  float     v8f;
typedef __attribute__((ext_vector_type(8)))  unsigned  uint8v;

__device__ __forceinline__ unsigned short f32_to_bf16(float f) {
  unsigned u = __builtin_bit_cast(unsigned, f);
  u += 0x7FFFu + ((u >> 16) & 1u);      // round-to-nearest-even
  return (unsigned short)(u >> 16);
}

// 16-element bf16 fragment from two 16B runs separated by `gap` elements.
__device__ __forceinline__ v16bf load_pair(const unsigned short* p, int gap) {
  uint4 lo = *(const uint4*)p;
  uint4 hi = *(const uint4*)(p + gap);
  uint8v u = {lo.x, lo.y, lo.z, lo.w, hi.x, hi.y, hi.z, hi.w};
  return __builtin_bit_cast(v16bf, u);
}

__device__ __forceinline__ float fast_sigmoid(float x) {
  return __builtin_amdgcn_rcpf(1.f + __expf(-x));
}
__device__ __forceinline__ float fast_tanh(float x) {
  return 2.f * __builtin_amdgcn_rcpf(1.f + __expf(-2.f * x)) - 1.f;
}

// ---------------- prep: W (D+H,4H) f32 -> Wt[4H][KPAD] bf16 (transposed, padded)
__global__ void prep_w(const float* __restrict__ W, unsigned short* __restrict__ Wt,
                       long total) {
  long idx = (long)blockIdx.x * blockDim.x + threadIdx.x;
  if (idx >= total) return;
  int n  = (int)(idx / KPAD);
  int kp = (int)(idx % KPAD);
  float v = 0.f;
  if (kp < D_)        v = W[(long)kp * FOURH + n];                 // x-weight rows 0..299
  else if (kp >= KXP) v = W[(long)(kp - (KXP - D_)) * FOURH + n];  // h-weight rows 300..2347
  Wt[idx] = f32_to_bf16(v);
}

// ---------------- prep: x (B,T,D) f32 -> Xs[T][B][KXP] bf16 (zero-padded)
__global__ void prep_x(const float* __restrict__ X, unsigned short* __restrict__ Xs) {
  int idx = blockIdx.x * blockDim.x + threadIdx.x;  // < T_*B_*KXP
  int k  = idx % KXP;
  int rb = idx / KXP;            // t*B_ + b
  int b  = rb % B_;
  int t  = rb / B_;
  float v = (k < D_) ? X[((long)b * T_ + t) * D_ + k] : 0.f;
  Xs[idx] = f32_to_bf16(v);
}

// ---------------- prep: h0 = 0 (bf16), c = 0 (f32)
__global__ void init_state(unsigned short* __restrict__ h0, float* __restrict__ c) {
  int idx = blockIdx.x * blockDim.x + threadIdx.x;
  if (idx < B_ * H_) { h0[idx] = 0; c[idx] = 0.f; }
}

// ---------------- one LSTM time step ----------------
// grid.x = H_/16 = 128 blocks (16 hidden units, all 4 gates); 256 threads = 8 waves.
// Wave w owns M-tile w (batch rows [16w,16w+16)) x 4 gate tiles -> 4 WMMA per K-chunk.
// B tiles staged once per block into a 3-deep LDS ring via async-to-LDS (ASYNCcnt):
// chunk c+2 is staged while chunk c computes, so each copy gets a full iteration
// of latency before s_wait_asynccnt(1) + barrier.
__global__ __launch_bounds__(256) void lstm_step(
    const unsigned short* __restrict__ Xs,     // [T][B][KXP] bf16
    const unsigned short* __restrict__ Wt,     // [4H][KPAD]  bf16
    const float*          __restrict__ bias,   // [4H] f32
    const unsigned short* __restrict__ hprev,  // [B][H] bf16
    unsigned short*       __restrict__ hnext,  // [B][H] bf16
    float*                __restrict__ cstate, // [B][H] f32
    float*                __restrict__ out,    // [B][T][H] f32
    int t) {
  __shared__ __align__(16) unsigned short Bs[3][4][16][BROW];
  constexpr unsigned BUFB = 4 * 16 * BROW * 2;   // 5120 bytes per ring slot

  const int tid   = threadIdx.x;
  const int lane  = tid & 31;
  const int wave  = tid >> 5;           // M-tile 0..7
  const int l15   = lane & 15;
  const int khalf = lane >> 4;
  const int colb  = blockIdx.x * 16;
  const int col   = colb + l15;         // hidden unit (B row / C col)

  // --- async B staging: exactly one 16B transfer per thread per chunk
  const int g_ = tid >> 6, r_ = (tid >> 2) & 15, q_ = tid & 3;
  const unsigned short* brow = Wt + (long)(g_ * H_ + colb + r_) * KPAD + q_ * 8;
  const unsigned ldsA = (unsigned)(size_t)&Bs[0][g_][r_][q_ * 8];

  // --- A source pointers (lane-half K interleave folded in)
  const int arow = wave * 16 + l15;
  const unsigned short* xr = Xs + ((long)t * B_ + arow) * KXP + khalf * 8;
  const unsigned short* hr = hprev + (long)arow * H_ + khalf * 8;

  // --- per-gate LDS fragment offsets (elements within one ring slot)
  const int f0 = (0 * 16 + l15) * BROW + khalf * 8;
  const int f1 = (1 * 16 + l15) * BROW + khalf * 8;
  const int f2 = (2 * 16 + l15) * BROW + khalf * 8;
  const int f3 = (3 * 16 + l15) * BROW + khalf * 8;

  v8f acc0 = {}, acc1 = {}, acc2 = {}, acc3 = {};

  // --- prologue: stage chunks 0 and 1, prefetch A chunk 0
  asm volatile("global_load_async_to_lds_b128 %0, %1, off"
               :: "v"(ldsA), "v"(brow) : "memory");
  asm volatile("global_load_async_to_lds_b128 %0, %1, off"
               :: "v"(ldsA + BUFB), "v"(brow + 32) : "memory");
  v16bf an = load_pair(xr, 16);
  asm volatile("s_wait_asynccnt 0x1" ::: "memory");  // chunk 0 arrived (in-order)
  __syncthreads();

  int      ri   = 0;         // ring slot holding chunk c
  unsigned soff = 2 * BUFB;  // ring-slot byte offset for staging chunk c+2

  for (int c = 0; c < NCHUNK; ++c) {
    const v16bf a = an;
    const int cn = c + 1, c2 = c + 2;
    if (c2 < NCHUNK)   // stage chunk c+2 (slot freed by barrier at end of iter c-1)
      asm volatile("global_load_async_to_lds_b128 %0, %1, off"
                   :: "v"(ldsA + soff), "v"(brow + c2 * 32) : "memory");
    if (cn < NCHUNK) { // prefetch next A fragment into registers
      const unsigned short* p = (cn < XCHUNK) ? (xr + cn * 32) : (hr + (cn - XCHUNK) * 32);
      an = load_pair(p, 16);
    }

    const unsigned short* bb = &Bs[ri][0][0][0];
    const v16bf b0 = load_pair(bb + f0, 16);
    const v16bf b1 = load_pair(bb + f1, 16);
    const v16bf b2 = load_pair(bb + f2, 16);
    const v16bf b3 = load_pair(bb + f3, 16);
    acc0 = __builtin_amdgcn_wmma_f32_16x16x32_bf16(false, a, false, b0, (short)0, acc0, false, false);
    acc1 = __builtin_amdgcn_wmma_f32_16x16x32_bf16(false, a, false, b1, (short)0, acc1, false, false);
    acc2 = __builtin_amdgcn_wmma_f32_16x16x32_bf16(false, a, false, b2, (short)0, acc2, false, false);
    acc3 = __builtin_amdgcn_wmma_f32_16x16x32_bf16(false, a, false, b3, (short)0, acc3, false, false);

    if (cn < NCHUNK) {
      if (c2 < NCHUNK)   // allow chunk c+2's copy to stay outstanding
        asm volatile("s_wait_asynccnt 0x1" ::: "memory");
      else
        asm volatile("s_wait_asynccnt 0x0" ::: "memory");
      __syncthreads();   // chunk c+1 visible; everyone done with slot ri
    }
    ri   = (ri == 2) ? 0 : ri + 1;
    soff = (soff == 2 * BUFB) ? 0u : soff + BUFB;
  }

  // --- gate math from accumulators (C layout: VGPR v -> row v + 8*khalf, col = l15)
  const float bi = bias[0 * H_ + col];
  const float bf = bias[1 * H_ + col];
  const float bg = bias[2 * H_ + col];
  const float bo = bias[3 * H_ + col];
#pragma unroll
  for (int v = 0; v < 8; ++v) {
    const int  row = wave * 16 + v + khalf * 8;
    const long idx = (long)row * H_ + col;
    const float ig = fast_sigmoid(acc0[v] + bi);
    const float fg = fast_sigmoid(acc1[v] + bf);
    const float gg = fast_tanh   (acc2[v] + bg);
    const float og = fast_sigmoid(acc3[v] + bo);
    const float cw = fg * cstate[idx] + ig * gg;
    cstate[idx] = cw;
    const float hn = og * fast_tanh(cw);
    out[((long)row * T_ + t) * H_ + col] = hn;   // outputs are (B,T,H)
    hnext[idx] = f32_to_bf16(hn);
  }
}

// ---------------- host ----------------
extern "C" void kernel_launch(void* const* d_in, const int* in_sizes, int n_in,
                              void* d_out, int out_size, void* d_ws, size_t ws_size,
                              hipStream_t stream) {
  const float* x    = (const float*)d_in[0];  // (B,T,D) f32
  // d_in[1] = lengths: unused (reference returns the full hidden sequence)
  const float* W    = (const float*)d_in[2];  // (D+H, 4H) f32
  const float* bias = (const float*)d_in[3];  // (4H) f32
  float*       out  = (float*)d_out;          // (B,T,H) f32

  char*  ws  = (char*)d_ws;
  size_t off = 0;
  const size_t xs_bytes = (size_t)T_ * B_ * KXP * 2;
  const size_t wt_bytes = (size_t)FOURH * KPAD * 2;
  const size_t h_bytes  = (size_t)B_ * H_ * 2;
  const size_t c_bytes  = (size_t)B_ * H_ * 4;

  unsigned short* Xs = (unsigned short*)(ws + off); off = (off + xs_bytes + 255) & ~(size_t)255;
  unsigned short* Wt = (unsigned short*)(ws + off); off = (off + wt_bytes + 255) & ~(size_t)255;
  unsigned short* h0 = (unsigned short*)(ws + off); off = (off + h_bytes  + 255) & ~(size_t)255;
  unsigned short* h1 = (unsigned short*)(ws + off); off = (off + h_bytes  + 255) & ~(size_t)255;
  float*          cs = (float*)(ws + off);          off = (off + c_bytes  + 255) & ~(size_t)255;

  {
    const int n = T_ * B_ * KXP;
    prep_x<<<(n + 255) / 256, 256, 0, stream>>>(x, Xs);
  }
  {
    const long n = (long)FOURH * KPAD;
    prep_w<<<(unsigned)((n + 255) / 256), 256, 0, stream>>>(W, Wt, n);
  }
  {
    const int n = B_ * H_;
    init_state<<<(n + 255) / 256, 256, 0, stream>>>(h0, cs);
  }

  unsigned short* hcur = h0;
  unsigned short* hnxt = h1;
  for (int t = 0; t < T_; ++t) {
    lstm_step<<<H_ / 16, 256, 0, stream>>>(Xs, Wt, bias, hcur, hnxt, cs, out, t);
    unsigned short* tmp = hcur; hcur = hnxt; hnxt = tmp;
  }
}
